// segmentation_model_67327907332570
// MI455X (gfx1250) — matrix operand order repaired and checked by reading, not compiled, not run on hardware
//
#include <hip/hip_runtime.h>
#include <hip/hip_bf16.h>
#include <stdint.h>

// ---------------- problem constants ----------------
#define BB      8
#define LL      1024
#define DM      256          // D_MODEL
#define DI      512          // D_INNER
#define DSTATE  16
#define DTR     16
#define BL      (BB*LL)      // 8192 rows

typedef __attribute__((ext_vector_type(16))) __bf16 bf16x16;
typedef __attribute__((ext_vector_type(8)))  __bf16 bf16x8;
typedef __attribute__((ext_vector_type(8)))  float  f32x8;

// ======================================================================
// bf16 WMMA GEMM:  C[M,N] = A[M,K] * W[N,K]^T   (f32 accumulate)
// One wave computes a 16 x (16*NT) strip of C, reusing the A fragment
// across NT WMMAs per k-step (compile-time NT -> fully unrolled, no
// divergent guards, deep s_clause load batches + partial loadcnt waits).
// Requires M%16==0, K%32==0, N == gridDim.x * 16*NT.
// Lane layouts used here (wave32; A/B k order is contiguous, k = 16*kh + s):
//   A (16x32 bf16): row = lane&15, kh = lane>>4; two b128 loads at
//                   K = k0 + 16*kh (VGPR0-3) and K = k0 + 16*kh + 8 (VGPR4-7)
//   B (32x16 bf16): col = lane&15; 16 contiguous K at k0 + 16*kh
//   C/D (16x16 f32): VGPR r -> M = 8*kh + r, N = lane&15
// ======================================================================
template <int NT>
__global__ __launch_bounds__(128)
void gemm_bf16_wmma(const __bf16* __restrict__ A,
                    const __bf16* __restrict__ W,
                    float* __restrict__ C,
                    int M, int N, int K)
{
    const int lane = threadIdx.x & 31;
    const int wave = threadIdx.x >> 5;
    const int m0 = (blockIdx.y * 4 + wave) * 16;
    const int n0 = blockIdx.x * (16 * NT);
    if (m0 >= M) return;

    const int row = lane & 15;
    const int kh  = lane >> 4;

    const __bf16* ap = A + (size_t)(m0 + row) * K;
    const __bf16* bp = W + (size_t)(n0 + row) * K + kh * 16;

    f32x8 acc[NT];
#pragma unroll
    for (int t = 0; t < NT; ++t) acc[t] = (f32x8){};

    for (int k0 = 0; k0 < K; k0 += 32) {
        bf16x8 alo = *reinterpret_cast<const bf16x8*>(ap + k0 + kh * 16);
        bf16x8 ahi = *reinterpret_cast<const bf16x8*>(ap + k0 + kh * 16 + 8);
        bf16x16 av;
#pragma unroll
        for (int i = 0; i < 8; ++i) { av[i] = alo[i]; av[8 + i] = ahi[i]; }
#pragma unroll
        for (int t = 0; t < NT; ++t) {
            bf16x16 bv = *reinterpret_cast<const bf16x16*>(
                bp + (size_t)t * 16 * K + k0);
            acc[t] = __builtin_amdgcn_wmma_f32_16x16x32_bf16(
                         false, av, false, bv, (short)0, acc[t],
                         false, false);
        }
    }

#pragma unroll
    for (int t = 0; t < NT; ++t) {
        float* cp = C + (size_t)(m0 + kh * 8) * N + n0 + t * 16 + row;
#pragma unroll
        for (int r = 0; r < 8; ++r) cp[(size_t)r * N] = acc[t][r];
    }
}

// ---------------- elementwise helpers ----------------
__global__ void cast_f32_bf16(const float* __restrict__ src,
                              __bf16* __restrict__ dst, int n)
{
    int i = blockIdx.x * blockDim.x + threadIdx.x;
    if (i < n) dst[i] = (__bf16)src[i];
}

// time-flip along L while casting: xr[b, L-1-l, :] = x[b, l, :]
__global__ void flip_cast_bf16(const float* __restrict__ x,
                               __bf16* __restrict__ xr)
{
    int i = blockIdx.x * blockDim.x + threadIdx.x;      // B*L*DM
    if (i >= BL * DM) return;
    int dd = i & (DM - 1);
    int m  = i >> 8;             // b*L + l
    int l  = m & (LL - 1);
    int b  = m >> 10;
    xr[((size_t)(b * LL + (LL - 1 - l)) * DM) + dd] = (__bf16)x[i];
}

// pad dt_w (512x16) -> bf16 (512x32), upper 16 K zero
__global__ void pack_dtw(const float* __restrict__ dtw,
                         __bf16* __restrict__ pad)
{
    int i = blockIdx.x * blockDim.x + threadIdx.x;      // 512*32
    if (i >= DI * 32) return;
    int j = i & 31, d = i >> 5;
    pad[i] = (j < DTR) ? (__bf16)dtw[d * DTR + j] : (__bf16)0.f;
}

// causal conv1d (D_CONV=4) + SiLU over the xc half of xz; emit f32 + bf16
__global__ void conv_silu(const float* __restrict__ xz,
                          const float* __restrict__ conv_w,
                          const float* __restrict__ conv_b,
                          float* __restrict__ xcf,
                          __bf16* __restrict__ xcbf)
{
    int idx = blockIdx.x * blockDim.x + threadIdx.x;    // B*L*DI
    if (idx >= BL * DI) return;
    int d = idx & (DI - 1);
    int m = idx >> 9;            // b*L + l
    int l = m & (LL - 1);
    float y = conv_b[d];
#pragma unroll
    for (int k = 0; k < 4; ++k) {
        int lp = l - 3 + k;
        if (lp >= 0)
            y += xz[(size_t)(m - l + lp) * (2 * DI) + d] * conv_w[d * 4 + k];
    }
    y = y / (1.f + __expf(-y));          // SiLU
    xcf[idx]  = y;
    xcbf[idx] = (__bf16)y;
}

// split proj (BLx48) -> dt (padded bf16 BLx32), Bm (BLx16 f32), Cm (BLx16 f32)
__global__ void split_proj(const float* __restrict__ proj,
                           __bf16* __restrict__ dtpad,
                           float* __restrict__ Bm,
                           float* __restrict__ Cm)
{
    int i = blockIdx.x * blockDim.x + threadIdx.x;      // BL*64
    if (i >= BL * 64) return;
    int t = i & 63, m = i >> 6;
    if (t < 32)
        dtpad[(size_t)m * 32 + t] = (t < DTR) ? (__bf16)proj[(size_t)m * 48 + t]
                                              : (__bf16)0.f;
    else if (t < 48)
        Bm[(size_t)m * DSTATE + (t - 32)] = proj[(size_t)m * 48 + (t - 16)];
    else
        Cm[(size_t)m * DSTATE + (t - 48)] = proj[(size_t)m * 48 + (t - 16)];
}

// ======================================================================
// Selective scan: 16 lanes per (b,d) channel, one lane per state n.
// h_n(l) = exp(dt*A_n)*h_n(l-1) + dt*xc*B_n ;  y = sum_n h_n*C_n
// then y = (y + xc*D) * silu(z)  -> bf16 for out_proj GEMM.
// 65536 threads total (2048 wave32s).
// ======================================================================
__global__ __launch_bounds__(256)
void selective_scan(const float* __restrict__ dtraw,   // BL x DI
                    const float* __restrict__ xcf,     // BL x DI
                    const float* __restrict__ xz,      // BL x 2DI (z at +DI)
                    const float* __restrict__ Bm,      // BL x 16
                    const float* __restrict__ Cm,      // BL x 16
                    const float* __restrict__ A_log,   // DI x 16
                    const float* __restrict__ dt_b,    // DI
                    const float* __restrict__ Dp,      // DI
                    __bf16* __restrict__ ybf)          // BL x DI
{
    int tid = blockIdx.x * blockDim.x + threadIdx.x;   // BB*DI*16
    if (tid >= BB * DI * DSTATE) return;
    int n = tid & 15;
    int g = tid >> 4;            // (b,d)
    int d = g & (DI - 1);
    int b = g >> 9;

    const float a   = -__expf(A_log[d * DSTATE + n]);
    const float dtb = dt_b[d];
    const float Dd  = Dp[d];

    float h = 0.f;
    for (int l = 0; l < LL; ++l) {
        size_t m = (size_t)(b * LL + l);
        float dtr = dtraw[m * DI + d] + dtb;
        float dt  = (dtr > 20.f) ? dtr : log1pf(__expf(dtr)); // softplus
        float xc  = xcf[m * DI + d];
        float Bn  = Bm[m * DSTATE + n];
        float Cn  = Cm[m * DSTATE + n];

        h = __expf(dt * a) * h + dt * xc * Bn;
        float p = h * Cn;
        // reduce over the 16-lane state group (stays inside wave32)
        p += __shfl_xor(p, 1, 32);
        p += __shfl_xor(p, 2, 32);
        p += __shfl_xor(p, 4, 32);
        p += __shfl_xor(p, 8, 32);
        if (n == 0) {
            float z  = xz[m * (2 * DI) + DI + d];
            float y  = (p + xc * Dd) * (z / (1.f + __expf(-z)));
            ybf[m * DI + d] = (__bf16)y;
        }
    }
}

// final: out[m,o] = sum_d (y1+y2)[m,d]*fc_w[o,d] + fc_b[o]
__global__ void final_fc(const float* __restrict__ y1,
                         const float* __restrict__ y2,
                         const float* __restrict__ fcw,
                         const float* __restrict__ fcb,
                         float* __restrict__ out)
{
    int m = blockIdx.x * blockDim.x + threadIdx.x;
    if (m >= BL) return;
    float a0 = 0.f, a1 = 0.f, a2 = 0.f;
    for (int dd = 0; dd < DM; ++dd) {
        float s = y1[(size_t)m * DM + dd] + y2[(size_t)m * DM + dd];
        a0 += s * fcw[dd];
        a1 += s * fcw[DM + dd];
        a2 += s * fcw[2 * DM + dd];
    }
    out[m * 3 + 0] = a0 + fcb[0];
    out[m * 3 + 1] = a1 + fcb[1];
    out[m * 3 + 2] = a2 + fcb[2];
}

// ======================================================================
// Host orchestration
// ======================================================================
static inline size_t align256(size_t x) { return (x + 255) & ~(size_t)255; }

extern "C" void kernel_launch(void* const* d_in, const int* in_sizes, int n_in,
                              void* d_out, int out_size, void* d_ws, size_t ws_size,
                              hipStream_t stream)
{
    (void)in_sizes; (void)n_in; (void)out_size; (void)ws_size;

    const float* x_in = (const float*)d_in[0];
    // per-block params: in_w, conv_w, conv_b, xproj_w, dt_w, dt_b, A_log, D, out_w
    const float* p[2][9];
    for (int s = 0; s < 2; ++s)
        for (int j = 0; j < 9; ++j)
            p[s][j] = (const float*)d_in[1 + s * 9 + j];
    const float* fc_w = (const float*)d_in[19];
    const float* fc_b = (const float*)d_in[20];
    float* out = (float*)d_out;

    // ---- workspace bump allocator ----
    char* ws = (char*)d_ws;
    size_t off = 0;
    auto alloc = [&](size_t bytes) -> char* {
        char* ptr = ws + off;
        off += align256(bytes);
        return ptr;
    };
    __bf16* xf_bf   = (__bf16*)alloc((size_t)BL * DM * 2);
    __bf16* xr_bf   = (__bf16*)alloc((size_t)BL * DM * 2);
    __bf16* w_in[2], *w_xp[2], *w_out[2], *w_dtw[2];
    for (int s = 0; s < 2; ++s) {
        w_in[s]  = (__bf16*)alloc((size_t)(2 * DI) * DM * 2);
        w_xp[s]  = (__bf16*)alloc((size_t)48 * DI * 2);
        w_out[s] = (__bf16*)alloc((size_t)DM * DI * 2);
        w_dtw[s] = (__bf16*)alloc((size_t)DI * 32 * 2);
    }
    float*  xz    = (float*) alloc((size_t)BL * (2 * DI) * 4);   // 32 MB
    float*  xcf   = (float*) alloc((size_t)BL * DI * 4);         // 16 MB
    __bf16* xcbf  = (__bf16*)alloc((size_t)BL * DI * 2);         //  8 MB
    float*  proj  = (float*) alloc((size_t)BL * 48 * 4);
    __bf16* dtpad = (__bf16*)alloc((size_t)BL * 32 * 2);
    float*  Bm    = (float*) alloc((size_t)BL * DSTATE * 4);
    float*  Cm    = (float*) alloc((size_t)BL * DSTATE * 4);
    float*  dtraw = (float*) alloc((size_t)BL * DI * 4);         // 16 MB
    __bf16* ybf   = (__bf16*)alloc((size_t)BL * DI * 2);         //  8 MB
    float*  ymm[2];
    ymm[0] = (float*)alloc((size_t)BL * DM * 4);
    ymm[1] = (float*)alloc((size_t)BL * DM * 4);

    const int T = 256;
    auto blocks = [&](int n) { return (n + T - 1) / T; };

    // ---- phase 0: casts / packs ----
    cast_f32_bf16<<<blocks(BL * DM), T, 0, stream>>>(x_in, xf_bf, BL * DM);
    flip_cast_bf16<<<blocks(BL * DM), T, 0, stream>>>(x_in, xr_bf);
    for (int s = 0; s < 2; ++s) {
        cast_f32_bf16<<<blocks(2 * DI * DM), T, 0, stream>>>(p[s][0], w_in[s], 2 * DI * DM);
        cast_f32_bf16<<<blocks(48 * DI), T, 0, stream>>>(p[s][3], w_xp[s], 48 * DI);
        cast_f32_bf16<<<blocks(DM * DI), T, 0, stream>>>(p[s][8], w_out[s], DM * DI);
        pack_dtw<<<blocks(DI * 32), T, 0, stream>>>(p[s][4], w_dtw[s]);
    }

    // ---- per-block pipeline ----
    for (int s = 0; s < 2; ++s) {
        const __bf16* xbf = (s == 0) ? xf_bf : xr_bf;

        // in_proj: [8192,256] x [1024,256]^T -> xz [8192,1024]  (N=1024: 16 strips)
        gemm_bf16_wmma<4><<<dim3((2 * DI) / 64, BL / 64), 128, 0, stream>>>(
            xbf, w_in[s], xz, BL, 2 * DI, DM);

        // conv1d + silu on xc half
        conv_silu<<<blocks(BL * DI), T, 0, stream>>>(xz, p[s][1], p[s][2], xcf, xcbf);

        // x_proj: [8192,512] x [48,512]^T -> proj [8192,48]  (N=48: one 3-tile strip)
        gemm_bf16_wmma<3><<<dim3(1, BL / 64), 128, 0, stream>>>(
            xcbf, w_xp[s], proj, BL, 48, DI);

        split_proj<<<blocks(BL * 64), T, 0, stream>>>(proj, dtpad, Bm, Cm);

        // dt_proj: [8192,32] x [512,32]^T -> dtraw [8192,512]
        gemm_bf16_wmma<4><<<dim3(DI / 64, BL / 64), 128, 0, stream>>>(
            dtpad, w_dtw[s], dtraw, BL, DI, 32);

        // selective scan + gating -> ybf
        selective_scan<<<(BB * DI * DSTATE) / 256, 256, 0, stream>>>(
            dtraw, xcf, xz, Bm, Cm, p[s][6], p[s][5], p[s][7], ybf);

        // out_proj: [8192,512] x [256,512]^T -> y_s [8192,256]
        gemm_bf16_wmma<4><<<dim3(DM / 64, BL / 64), 128, 0, stream>>>(
            ybf, w_out[s], ymm[s], BL, DM, DI);
    }

    // ---- final FC (y1 + y2) -> [8192,3] ----
    final_fc<<<blocks(BL), T, 0, stream>>>(ymm[0], ymm[1], fc_w, fc_b, out);
}
